// GatedAttention_19902878450126
// MI455X (gfx1250) — matrix operand order repaired and checked by reading, not compiled, not run on hardware
//
#include <hip/hip_runtime.h>

#define BQ  4
#define SEQ 4096
#define HID 1024
#define ATT 512

typedef __attribute__((ext_vector_type(16))) __bf16   v16bf;
typedef __attribute__((ext_vector_type(8)))  float    v8f;
typedef __attribute__((ext_vector_type(4)))  unsigned u32x4;
typedef __attribute__((ext_vector_type(8)))  unsigned u32x8;
typedef __attribute__((ext_vector_type(2)))  unsigned u32x2;
typedef __attribute__((ext_vector_type(4)))  float    f32x4;

union AF { v16bf v; u32x4 q[2]; };
union BF { v16bf v; u32x4 q[2]; unsigned u[8]; };

__device__ __forceinline__ v8f wmma_bf16(v16bf a, v16bf b, v8f c) {
  return __builtin_amdgcn_wmma_f32_16x16x32_bf16(false, a, false, b, (short)0, c,
                                                 false, false);
}

// ---- TDM: async 2-D tile DMA global->LDS (64 rows x 256 DWORDs, +4 DW pad/row)
// D# per CDNA5 ISA 8.3/8.4: group0 {count|lds_addr|global_addr|type=2},
// group1 {data_size=4B, pad_enable, pad_interval=256DW, pad_amount=4DW,
//         tensor 256x4096, tile 256x64, stride 256}; groups 2/3 zero (2-D).
__device__ __forceinline__ void tdm_load_chunk(unsigned long long ga, unsigned ldsoff) {
  u32x4 g0; u32x8 g1; u32x4 gz;
  g0[0] = 1u;                                            // count=1 (user D#)
  g0[1] = ldsoff;                                        // lds_addr (bytes)
  g0[2] = (unsigned)ga;                                  // global_addr[31:0]
  g0[3] = (unsigned)((ga >> 32) & 0x01FFFFFFull) | (2u << 30);  // [56:32]|type=2
  g1[0] = (2u << 16) | (1u << 20) | (7u << 22) | (3u << 25);
  g1[1] = (256u & 0xFFFFu) << 16;                        // tensor_dim0 lo16
  g1[2] = (256u >> 16) | ((4096u & 0xFFFFu) << 16);      // dim0 hi | dim1 lo
  g1[3] = (4096u >> 16) | (256u << 16);                  // dim1 hi | tile_dim0
  g1[4] = 64u;                                           // tile_dim1 (tile_dim2=0)
  g1[5] = 256u;                                          // tensor_dim0_stride lo32
  g1[6] = 0u;
  g1[7] = 0u;
  gz[0] = 0u; gz[1] = 0u; gz[2] = 0u; gz[3] = 0u;
  asm volatile("tensor_load_to_lds %0, %1, %2, %3"
               :: "s"(g0), "s"(g1), "s"(gz), "s"(gz) : "memory");
}

// ---------------- prep: f32 -> bf16 conversions ----------------
__global__ void __launch_bounds__(256)
k_prep_x(const float* __restrict__ x, __bf16* __restrict__ xb) {
  int idx = (blockIdx.x * 256 + threadIdx.x) * 4;
  f32x4 f = *(const f32x4*)(x + idx);
  union { __bf16 h[4]; u32x2 u; } p;
  p.h[0] = (__bf16)f.x; p.h[1] = (__bf16)f.y;
  p.h[2] = (__bf16)f.z; p.h[3] = (__bf16)f.w;
  *(u32x2*)(xb + idx) = p.u;
}

// W[H][A] f32 -> col-major pair-interleaved: WP[a*512 + k2] = {W[2k2][a], W[2k2+1][a]}
__global__ void __launch_bounds__(256)
k_prep_w(const float* __restrict__ W, unsigned* __restrict__ WP) {
  int t = blockIdx.x * 256 + threadIdx.x;     // t < ATT*(HID/2)
  int k2 = t & 511, a = t >> 9;
  union { __bf16 h[2]; unsigned u; } p;
  p.h[0] = (__bf16)W[(2 * k2) * ATT + a];
  p.h[1] = (__bf16)W[(2 * k2 + 1) * ATT + a];
  WP[t] = p.u;
}

// ---------------- pass 1: Qw = x@Wq+bq, Kw = x@Wk+bk ----------------
__global__ void __launch_bounds__(256)
k_proj(const __bf16* __restrict__ xb, const unsigned* __restrict__ wqP,
       const unsigned* __restrict__ wkP, const float* __restrict__ bq,
       const float* __restrict__ bk, __bf16* __restrict__ QwB,
       __bf16* __restrict__ KwB, unsigned* __restrict__ QwP,
       unsigned* __restrict__ KwP) {
  extern __shared__ char smem[];
  __bf16*   xT  = (__bf16*)smem;               // [128][72] bf16
  unsigned* wqT = (unsigned*)(smem + 18432);   // [128 a][36 k2] u32
  unsigned* wkT = (unsigned*)(smem + 36864);   // [128 a][36 k2] u32
  __bf16*   rp  = (__bf16*)(smem + 55296);     // [128][136] repack buffer

  const int b = blockIdx.z, i0 = blockIdx.y * 128, a0 = blockIdx.x * 128;
  const int tid = threadIdx.x;
  const int w = tid >> 5, L = tid & 31, half = L >> 4, l15 = L & 15;
  const int wi = (w & 3) * 32, wa = (w >> 2) * 64;

  const __bf16* xbase = xb + ((size_t)b * SEQ + i0) * HID;

  v8f accQ[2][4], accK[2][4];
#pragma unroll
  for (int mi = 0; mi < 2; ++mi)
#pragma unroll
    for (int na = 0; na < 4; ++na)
#pragma unroll
      for (int e = 0; e < 8; ++e) { accQ[mi][na][e] = 0.f; accK[mi][na][e] = 0.f; }

  for (int kt = 0; kt < HID / 64; ++kt) {
    __syncthreads();
#pragma unroll
    for (int s = 0; s < 4; ++s) {             // x tile 128x64 bf16
      int c = tid + s * 256, row = c >> 3, col = (c & 7) * 8;
      *(u32x4*)&xT[row * 72 + col] =
          *(const u32x4*)&xbase[row * HID + kt * 64 + col];
    }
#pragma unroll
    for (int s = 0; s < 4; ++s) {             // weight chunks 128a x 32 k2
      int c = tid + s * 256, a = c >> 3, seg = (c & 7) * 4;
      *(u32x4*)&wqT[a * 36 + seg] =
          *(const u32x4*)&wqP[(size_t)(a0 + a) * 512 + kt * 32 + seg];
      *(u32x4*)&wkT[a * 36 + seg] =
          *(const u32x4*)&wkP[(size_t)(a0 + a) * 512 + kt * 32 + seg];
    }
    __syncthreads();
    if (kt + 1 < HID / 64)
      __builtin_prefetch(&xbase[(tid >> 1) * HID + (kt + 1) * 64 + (tid & 1) * 32], 0, 1);
#pragma unroll
    for (int ks = 0; ks < 2; ++ks) {
      AF af[2];
#pragma unroll
      for (int mi = 0; mi < 2; ++mi) {
        const __bf16* p = &xT[(wi + 16 * mi + l15) * 72 + ks * 32 + 16 * half];
        af[mi].q[0] = *(const u32x4*)p;
        af[mi].q[1] = *(const u32x4*)(p + 8);
      }
#pragma unroll
      for (int na = 0; na < 4; ++na) {
        int aL = wa + 16 * na + l15;
        BF bfq, bfk;
        const unsigned* pq = &wqT[aL * 36 + ks * 16 + 8 * half];
        bfq.q[0] = *(const u32x4*)pq; bfq.q[1] = *(const u32x4*)(pq + 4);
        const unsigned* pk = &wkT[aL * 36 + ks * 16 + 8 * half];
        bfk.q[0] = *(const u32x4*)pk; bfk.q[1] = *(const u32x4*)(pk + 4);
#pragma unroll
        for (int mi = 0; mi < 2; ++mi) {
          accQ[mi][na] = wmma_bf16(af[mi].v, bfq.v, accQ[mi][na]);
          accK[mi][na] = wmma_bf16(af[mi].v, bfk.v, accK[mi][na]);
        }
      }
    }
  }
#pragma unroll
  for (int na = 0; na < 4; ++na) {            // bias epilogue
    float vq = bq[a0 + wa + 16 * na + l15];
    float vk = bk[a0 + wa + 16 * na + l15];
#pragma unroll
    for (int mi = 0; mi < 2; ++mi)
#pragma unroll
      for (int e = 0; e < 8; ++e) { accQ[mi][na][e] += vq; accK[mi][na][e] += vk; }
  }

  // ---- Q tile: repack through LDS, emit row-major + col-major-pairs ----
  __syncthreads();
#pragma unroll
  for (int mi = 0; mi < 2; ++mi)
#pragma unroll
    for (int na = 0; na < 4; ++na)
#pragma unroll
      for (int e = 0; e < 8; ++e)
        rp[(wi + 16 * mi + e + 8 * half) * 136 + wa + 16 * na + l15] =
            (__bf16)accQ[mi][na][e];
  __syncthreads();
  {
    __bf16* gb = QwB + ((size_t)b * SEQ + i0) * ATT + a0;
#pragma unroll
    for (int s = 0; s < 8; ++s) {
      int c = tid + s * 256, row = c >> 4, col = (c & 15) * 8;
      *(u32x4*)&gb[row * ATT + col] = *(const u32x4*)&rp[row * 136 + col];
    }
    unsigned* gp = QwP + ((size_t)b * SEQ + i0) * 256 + (a0 / 2);
#pragma unroll
    for (int s = 0; s < 32; ++s) {
      int c = tid + s * 256, il = c >> 6, k2l = c & 63;
      gp[(size_t)il * 256 + k2l] = *(const unsigned*)&rp[il * 136 + 2 * k2l];
    }
  }
  // ---- K tile ----
  __syncthreads();
#pragma unroll
  for (int mi = 0; mi < 2; ++mi)
#pragma unroll
    for (int na = 0; na < 4; ++na)
#pragma unroll
      for (int e = 0; e < 8; ++e)
        rp[(wi + 16 * mi + e + 8 * half) * 136 + wa + 16 * na + l15] =
            (__bf16)accK[mi][na][e];
  __syncthreads();
  {
    __bf16* gb = KwB + ((size_t)b * SEQ + i0) * ATT + a0;
#pragma unroll
    for (int s = 0; s < 8; ++s) {
      int c = tid + s * 256, row = c >> 4, col = (c & 15) * 8;
      *(u32x4*)&gb[row * ATT + col] = *(const u32x4*)&rp[row * 136 + col];
    }
    unsigned* gp = KwP + ((size_t)b * SEQ + i0) * 256 + (a0 / 2);
#pragma unroll
    for (int s = 0; s < 32; ++s) {
      int c = tid + s * 256, il = c >> 6, k2l = c & 63;
      gp[(size_t)il * 256 + k2l] = *(const unsigned*)&rp[il * 136 + 2 * k2l];
    }
  }
}

// ---------------- gate: g = sigmoid(Qw @ Wv + bv) ----------------
__global__ void __launch_bounds__(256)
k_gate(const __bf16* __restrict__ QwB, const float* __restrict__ Wv,
       const float* __restrict__ bv, float* __restrict__ g) {
  const int row = blockIdx.x * 8 + (threadIdx.x >> 5);
  const int lane = threadIdx.x & 31;
  const __bf16* q = QwB + (size_t)row * ATT;
  float sum = 0.f;
#pragma unroll
  for (int c = 0; c < ATT / 32; ++c)
    sum += (float)q[lane + c * 32] * Wv[lane + c * 32];
#pragma unroll
  for (int o = 16; o; o >>= 1) sum += __shfl_down(sum, o, 32);
  if (lane == 0) g[row] = 1.f / (1.f + __expf(-(sum + bv[0])));
}

// ---------------- pass 2: per-column softmax stats (online, TDM-fed) ----------------
// T[j,i] = dot(Kw[j], Qw[i]) = scores[i,j]; reduce over i per row j.
__global__ void __launch_bounds__(256)
k_stats(const __bf16* __restrict__ KwB, const unsigned* __restrict__ QwP,
        float* __restrict__ colmax, float* __restrict__ colsum) {
  extern __shared__ char smem[];
  __bf16*   kT  = (__bf16*)smem;                 // [64][520] bf16
  unsigned* qB0 = (unsigned*)(smem + 66560);     // [64][260] u32 (TDM dest 0)
  unsigned* qB1 = (unsigned*)(smem + 133120);    // [64][260] u32 (TDM dest 1)
  float* m_run  = (float*)(smem + 199680);       // [64][2]
  float* s_run  = (float*)(smem + 200192);       // [64][2]

  const int b = blockIdx.y, j0 = blockIdx.x * 64;
  const int tid = threadIdx.x;
  const int L = tid & 31, half = L >> 4, l15 = L & 15;
  const int w = tid >> 5, wy = w >> 1, wx = w & 1, jw = wy * 16;

  const __bf16* kb = KwB + ((size_t)b * SEQ + j0) * ATT;
#pragma unroll
  for (int s = 0; s < 16; ++s) {                 // Kw tile resident all sweep
    int c = tid + s * 256, row = c >> 6, col = (c & 63) * 8;
    *(u32x4*)&kT[row * 520 + col] = *(const u32x4*)&kb[row * ATT + col];
  }
  if (tid < 128) { m_run[tid] = -1e30f; s_run[tid] = 0.f; }
  __syncthreads();

  AF afx[16];                                    // hoist all A-fragments
#pragma unroll
  for (int ks = 0; ks < 16; ++ks) {
    const __bf16* p = &kT[(jw + l15) * 520 + ks * 32 + 16 * half];
    afx[ks].q[0] = *(const u32x4*)p;
    afx[ks].q[1] = *(const u32x4*)(p + 8);
  }

  const unsigned long long gbase =
      (unsigned long long)(const void*)(QwP + (size_t)b * SEQ * 256);
  if (tid < 32) tdm_load_chunk(gbase, 66560u);

  for (int ic = 0; ic < SEQ / 64; ++ic) {
    const int i0 = ic * 64;
    __syncthreads();                             // buffer (ic+1)&1 is free
    if (tid < 32) {
      if (ic + 1 < SEQ / 64) {
        tdm_load_chunk(gbase + (unsigned long long)(ic + 1) * 64u * 1024u,
                       66560u + (unsigned)((ic + 1) & 1) * 66560u);
        __builtin_amdgcn_s_wait_tensorcnt(1);
      } else {
        __builtin_amdgcn_s_wait_tensorcnt(0);
      }
    }
    __syncthreads();                             // buffer ic&1 ready
    const unsigned* qb = (ic & 1) ? qB1 : qB0;

    v8f acc[2];
#pragma unroll
    for (int nt = 0; nt < 2; ++nt)
#pragma unroll
      for (int e = 0; e < 8; ++e) acc[nt][e] = 0.f;
#pragma unroll
    for (int ks = 0; ks < 16; ++ks) {
#pragma unroll
      for (int nt = 0; nt < 2; ++nt) {
        BF bb;
        const unsigned* q = &qb[(wx * 32 + nt * 16 + l15) * 260 + ks * 16 + 8 * half];
        bb.q[0] = *(const u32x4*)q;
        bb.q[1] = *(const u32x4*)(q + 4);
        acc[nt] = wmma_bf16(afx[ks].v, bb.v, acc[nt]);
      }
    }
    // online (max,sum); (row, slot wx) are wave-private
#pragma unroll
    for (int v = 0; v < 8; ++v) {
      const int row = jw + v + 8 * half;
      const int jg = j0 + row;
      float vals[2], rm = -1e30f;
#pragma unroll
      for (int nt = 0; nt < 2; ++nt) {
        int ig = i0 + wx * 32 + nt * 16 + l15;
        vals[nt] = (ig == jg) ? -1e30f : acc[nt][v];   // diagonal mask
        rm = fmaxf(rm, vals[nt]);
      }
#pragma unroll
      for (int d = 8; d; d >>= 1) rm = fmaxf(rm, __shfl_xor(rm, d, 32));
      const float m_old = m_run[row * 2 + wx];
      const float m_new = fmaxf(m_old, rm);
      float es = __expf(vals[0] - m_new) + __expf(vals[1] - m_new);
#pragma unroll
      for (int d = 8; d; d >>= 1) es += __shfl_xor(es, d, 32);
      if (l15 == 0) {
        s_run[row * 2 + wx] = s_run[row * 2 + wx] * __expf(m_old - m_new) + es;
        m_run[row * 2 + wx] = m_new;
      }
    }
  }
  __syncthreads();
  if (tid < 64) {
    float m0 = m_run[tid * 2], m1 = m_run[tid * 2 + 1];
    float s0 = s_run[tid * 2], s1 = s_run[tid * 2 + 1];
    float M = fmaxf(m0, m1);
    float S = s0 * __expf(m0 - M) + s1 * __expf(m1 - M);
    colmax[b * SEQ + j0 + tid] = M;
    colsum[b * SEQ + j0 + tid] = S;
  }
}

// ---------------- pass 3: recompute scores, softmax + gate, write out ----------------
__global__ void __launch_bounds__(256)
k_out(const __bf16* __restrict__ QwB, const unsigned* __restrict__ KwP,
      const float* __restrict__ gArr, const float* __restrict__ colmax,
      const float* __restrict__ colsum, float* __restrict__ out) {
  extern __shared__ char smem[];
  __bf16*   qT  = (__bf16*)smem;                 // [128][520] bf16
  unsigned* kB0 = (unsigned*)(smem + 133120);    // [64][260] u32 (TDM dest 0)
  unsigned* kB1 = (unsigned*)(smem + 199680);    // [64][260] u32 (TDM dest 1)
  float* gI = (float*)(smem + 266240);           // [128]
  float* mJ = (float*)(smem + 266752);           // [64]
  float* sJ = (float*)(smem + 267008);           // [64]

  const int b = blockIdx.z, i0 = blockIdx.y * 128, jb0 = blockIdx.x * 512;
  const int tid = threadIdx.x;
  const int L = tid & 31, half = L >> 4, l15 = L & 15;
  const int w = tid >> 5;
  const int wi = (w & 3) * 32, wj = (w >> 2) * 32;

  const __bf16* qb = QwB + ((size_t)b * SEQ + i0) * ATT;
#pragma unroll
  for (int s = 0; s < 32; ++s) {                 // Qw tile resident for 8 j-chunks
    int c = tid + s * 256, row = c >> 6, col = (c & 63) * 8;
    *(u32x4*)&qT[row * 520 + col] = *(const u32x4*)&qb[row * ATT + col];
  }
  if (tid < 128) gI[tid] = gArr[b * SEQ + i0 + tid];

  const unsigned long long gbase =
      (unsigned long long)(const void*)(KwP + (size_t)b * SEQ * 256) +
      (unsigned long long)jb0 * 1024u;
  if (tid < 32) tdm_load_chunk(gbase, 133120u);

  float* ob = out + ((size_t)b * SEQ + i0) * SEQ;

  for (int jc = 0; jc < 8; ++jc) {
    const int j0 = jb0 + jc * 64;
    __syncthreads();                             // buffer (jc+1)&1 is free
    if (tid < 32) {
      if (jc + 1 < 8) {
        tdm_load_chunk(gbase + (unsigned long long)(jc + 1) * 64u * 1024u,
                       133120u + (unsigned)((jc + 1) & 1) * 66560u);
        __builtin_amdgcn_s_wait_tensorcnt(1);
      } else {
        __builtin_amdgcn_s_wait_tensorcnt(0);
      }
    }
    if (tid >= 192) {                            // waves 6-7 stage per-j stats
      int t = tid - 192;
      mJ[t] = colmax[b * SEQ + j0 + t];
      sJ[t] = 1.f / colsum[b * SEQ + j0 + t];
    }
    __syncthreads();                             // buffer jc&1 + stats ready
    const unsigned* kb2 = (jc & 1) ? kB1 : kB0;

    v8f acc[2][2];
#pragma unroll
    for (int mi = 0; mi < 2; ++mi)
#pragma unroll
      for (int nj = 0; nj < 2; ++nj)
#pragma unroll
        for (int e = 0; e < 8; ++e) acc[mi][nj][e] = 0.f;
#pragma unroll
    for (int ks = 0; ks < 16; ++ks) {
      AF af[2];
#pragma unroll
      for (int mi = 0; mi < 2; ++mi) {
        const __bf16* p = &qT[(wi + 16 * mi + l15) * 520 + ks * 32 + 16 * half];
        af[mi].q[0] = *(const u32x4*)p;
        af[mi].q[1] = *(const u32x4*)(p + 8);
      }
#pragma unroll
      for (int nj = 0; nj < 2; ++nj) {
        BF bb;
        const unsigned* q = &kb2[(wj + nj * 16 + l15) * 260 + ks * 16 + 8 * half];
        bb.q[0] = *(const u32x4*)q;
        bb.q[1] = *(const u32x4*)(q + 4);
#pragma unroll
        for (int mi = 0; mi < 2; ++mi)
          acc[mi][nj] = wmma_bf16(af[mi].v, bb.v, acc[mi][nj]);
      }
    }
#pragma unroll
    for (int mi = 0; mi < 2; ++mi)
#pragma unroll
      for (int nj = 0; nj < 2; ++nj) {
        const int jL = wj + 16 * nj + l15, jg = j0 + jL;
        const float mj = mJ[jL], isj = sJ[jL];
#pragma unroll
        for (int e = 0; e < 8; ++e) {
          const int iloc = wi + 16 * mi + e + 8 * half;
          const int ig = i0 + iloc;
          const float gi = gI[iloc];
          float o = (ig == jg) ? gi
                    : (1.f - gi) * __expf(acc[mi][nj][e] - mj) * isj;
          ob[(size_t)iloc * SEQ + jg] = o;
        }
      }
  }
}

// ---------------- host launcher ----------------
extern "C" void kernel_launch(void* const* d_in, const int* in_sizes, int n_in,
                              void* d_out, int out_size, void* d_ws, size_t ws_size,
                              hipStream_t stream) {
  (void)in_sizes; (void)n_in; (void)out_size; (void)ws_size;
  const float* x  = (const float*)d_in[0];
  const float* Wq = (const float*)d_in[1];
  const float* bq = (const float*)d_in[2];
  const float* Wk = (const float*)d_in[3];
  const float* bk = (const float*)d_in[4];
  const float* Wv = (const float*)d_in[5];
  const float* bv = (const float*)d_in[6];
  float* out = (float*)d_out;
  char* ws = (char*)d_ws;

  size_t off = 0;
  __bf16* xB = (__bf16*)(ws + off);      off += (size_t)BQ * SEQ * HID * 2;
  unsigned* wqP = (unsigned*)(ws + off); off += (size_t)(HID / 2) * ATT * 4;
  unsigned* wkP = (unsigned*)(ws + off); off += (size_t)(HID / 2) * ATT * 4;
  __bf16* QwB = (__bf16*)(ws + off);     off += (size_t)BQ * SEQ * ATT * 2;
  __bf16* KwB = (__bf16*)(ws + off);     off += (size_t)BQ * SEQ * ATT * 2;
  unsigned* QwP = (unsigned*)(ws + off); off += (size_t)BQ * SEQ * (ATT / 2) * 4;
  unsigned* KwP = (unsigned*)(ws + off); off += (size_t)BQ * SEQ * (ATT / 2) * 4;
  float* gArr = (float*)(ws + off);      off += (size_t)BQ * SEQ * 4;
  float* cm = (float*)(ws + off);        off += (size_t)BQ * SEQ * 4;
  float* cs = (float*)(ws + off);        off += (size_t)BQ * SEQ * 4;

  k_prep_x<<<(BQ * SEQ * HID) / (256 * 4), 256, 0, stream>>>(x, xB);
  k_prep_w<<<((HID / 2) * ATT) / 256, 256, 0, stream>>>(Wq, wqP);
  k_prep_w<<<((HID / 2) * ATT) / 256, 256, 0, stream>>>(Wk, wkP);
  k_proj<<<dim3(ATT / 128, SEQ / 128, BQ), 256, 90112, stream>>>(
      xB, wqP, wkP, bq, bk, QwB, KwB, QwP, KwP);
  k_gate<<<(BQ * SEQ) / 8, 256, 0, stream>>>(QwB, Wv, bv, gArr);
  k_stats<<<dim3(SEQ / 64, BQ), 256, 200704, stream>>>(KwB, QwP, cm, cs);
  k_out<<<dim3(SEQ / 512, SEQ / 128, BQ), 256, 267264, stream>>>(
      QwB, KwP, gArr, cm, cs, out);
}
